// SlowFastSSMM_24197845746022
// MI455X (gfx1250) — matrix operand order repaired and checked by reading, not compiled, not run on hardware
//
#include <hip/hip_runtime.h>
#include <hip/hip_bf16.h>
#include <math.h>

// ---------------------------------------------------------------------------
// SlowFastSSM for MI455X (gfx1250, wave32, WMMA bf16 16x16x32, async-to-LDS)
// ---------------------------------------------------------------------------

typedef __attribute__((ext_vector_type(16))) __bf16 v16bf;
typedef __attribute__((ext_vector_type(8)))  __bf16 v8bf;
typedef __attribute__((ext_vector_type(8)))  float  v8f;
typedef __attribute__((ext_vector_type(4)))  int    v4i;

typedef __attribute__((address_space(1))) v4i* gptr_v4i;   // global
typedef __attribute__((address_space(3))) v4i* lptr_v4i;   // LDS

#define WMMA_BF16(a, b, c) \
  __builtin_amdgcn_wmma_f32_16x16x32_bf16(false, (a), false, (b), (short)0, (c), false, false)

static constexpr int LF    = 32;
static constexpr int HOPF  = 16;
static constexpr int LS    = 96;
static constexpr int DELTA = 3;
static constexpr int H     = 128;
static constexpr int GH    = 256;
static constexpr int B     = 16;
static constexpr int T     = 65568;
static constexpr int NF    = (T - LF) / HOPF + 1;     // 4097
static constexpr int HOPS  = DELTA * HOPF;            // 48
static constexpr int NS    = (T - LS) / HOPS + 1;     // 1365
static constexpr int TOUT  = (NF - 1) * HOPF + LF;    // 65568
static constexpr int G3    = 3 * GH;                  // 768

static __device__ __forceinline__ __bf16 bfc(float f) { return (__bf16)f; }
static __device__ __forceinline__ float  sigf(float v) { return 1.0f / (1.0f + __expf(-v)); }

// ---------------------------------------------------------------------------
// Async global->LDS copy (gfx1250). Guarded so the build can never break.
// ---------------------------------------------------------------------------
#if defined(__has_builtin)
#if __has_builtin(__builtin_amdgcn_global_load_async_to_lds_b128)
#define HAVE_ASYNC_LDS 1
#endif
#endif

// Copy B*G3 = 12288 floats (49152 B) with 512 threads: 6 x b128 per thread.
static __device__ __forceinline__ void prefetch_xp(const float* __restrict__ src,
                                                   float* __restrict__ dst_lds,
                                                   int tid) {
#ifdef HAVE_ASYNC_LDS
#pragma unroll
  for (int i = 0; i < 6; ++i) {
    const int idx = (tid + i * 512) * 4;
    __builtin_amdgcn_global_load_async_to_lds_b128(
        (gptr_v4i)(src + idx), (lptr_v4i)(dst_lds + idx), 0, 0);
  }
#else
#pragma unroll
  for (int i = 0; i < 6; ++i) {
    const int idx = (tid + i * 512) * 4;
    *(float4*)(dst_lds + idx) = *(const float4*)(src + idx);
  }
#endif
}

static __device__ __forceinline__ void wait_async_keep6() {
#ifdef HAVE_ASYNC_LDS
#if __has_builtin(__builtin_amdgcn_s_wait_asynccnt)
  __builtin_amdgcn_s_wait_asynccnt(6);
#else
  asm volatile("s_wait_asynccnt 6" ::: "memory");
#endif
#endif
}

// ---------------------------------------------------------------------------
// B fragment (32x16, K x N): lane l -> column n = base_n + (l&15),
// k = k0 + ((l>=16)?16:0) + j, j=0..15. Packed form: 16 bf16 (32 B) per lane,
// fragments contiguous -> two global_load_b128 per fragment.
// ---------------------------------------------------------------------------
static __device__ __forceinline__ v16bf load_b_packed(const __bf16* __restrict__ base,
                                                      int frag, int lane) {
  return *(const v16bf*)(base + (((size_t)frag * 32 + lane) << 4));
}

// On-the-fly fp32 B fragment (used only in the fully parallel k1).
static __device__ __forceinline__ v16bf load_b_frag(const float* __restrict__ W,
                                                    int K, int base_n, int k0, int lane) {
  const float* p = W + (size_t)(base_n + (lane & 15)) * K + (k0 + ((lane >> 4) << 4));
  v16bf b;
#pragma unroll
  for (int j = 0; j < 16; j += 4) {
    float4 v = *(const float4*)(p + j);
    b[j + 0] = bfc(v.x); b[j + 1] = bfc(v.y);
    b[j + 2] = bfc(v.z); b[j + 3] = bfc(v.w);
  }
  return b;
}

// A fragment (16x32): lane l -> row m = l&15, half = l>>4;
//   a[0..7]  = k0 + half*16 + j ; a[8..15] = k0 + half*16 + 8 + j
static __device__ __forceinline__ v16bf load_a_row_f32(const float* __restrict__ row,
                                                       int k0, int half) {
  v16bf a;
  const float* p0 = row + k0 + half * 16;
  const float* p1 = row + k0 + half * 16 + 8;
#pragma unroll
  for (int j = 0; j < 8; ++j) a[j]     = bfc(p0[j]);
#pragma unroll
  for (int j = 0; j < 8; ++j) a[8 + j] = bfc(p1[j]);
  return a;
}

// A fragment from a bf16 row in LDS: two 16-byte loads + concat.
static __device__ __forceinline__ v16bf load_a_row_bf(const __bf16* __restrict__ row,
                                                      int k0, int half) {
  v8bf lo = *(const v8bf*)(row + k0 + half * 16);
  v8bf hi = *(const v8bf*)(row + k0 + half * 16 + 8);
  return __builtin_shufflevector(lo, hi, 0, 1, 2, 3, 4, 5, 6, 7,
                                 8, 9, 10, 11, 12, 13, 14, 15);
}

// ---------------------------------------------------------------------------
// Kernel 0: pack row-major fp32 weight W[N][K] into bf16 WMMA B-fragments.
// One thread per (fragment, lane); fragment = ntile*(K/32) + kc.
// ---------------------------------------------------------------------------
__global__ void k0_pack_wfrag(const float* __restrict__ W, int K,
                              __bf16* __restrict__ out, int nfrag) {
  const int g = blockIdx.x * 256 + threadIdx.x;
  if (g >= nfrag * 32) return;
  const int frag = g >> 5, lane = g & 31;
  const int kch  = K >> 5;
  const int nt   = frag / kch, kc = frag % kch;
  const float* p = W + (size_t)(nt * 16 + (lane & 15)) * K + kc * 32 + ((lane >> 4) << 4);
  __bf16* o = out + ((size_t)g << 4);
#pragma unroll
  for (int j = 0; j < 16; j += 4) {
    float4 v = *(const float4*)(p + j);
    o[j + 0] = bfc(v.x); o[j + 1] = bfc(v.y);
    o[j + 2] = bfc(v.z); o[j + 3] = bfc(v.w);
  }
}

// ---------------------------------------------------------------------------
// Kernel 1: x_proj[s][b][0..767] = slow_frames[b][s] @ W_ih^T + b_ih
// Grid: (6, NS), 8 waves/block, 1 wave = one 16x16 N-tile (K = 96).
// ---------------------------------------------------------------------------
__global__ void k1_slow_proj(const float* __restrict__ x,
                             const float* __restrict__ W_ih,
                             const float* __restrict__ b_ih,
                             float* __restrict__ xproj) {
  const int s     = blockIdx.y;
  const int ntile = (blockIdx.x << 3) + (threadIdx.x >> 5);  // 0..47
  const int lane  = threadIdx.x & 31;
  const int nl    = lane & 15;
  const int half  = lane >> 4;

  const float* xrow = x + (size_t)nl * T + (size_t)s * HOPS;

  v8f acc = {};
#pragma unroll
  for (int kc = 0; kc < 3; ++kc) {
    v16bf a  = load_a_row_f32(xrow, kc * 32, half);
    v16bf bb = load_b_frag(W_ih, LS, ntile * 16, kc * 32, lane);
    acc = WMMA_BF16(a, bb, acc);
  }

  const int   n    = ntile * 16 + nl;
  const float bias = b_ih[n];
  float* orow = xproj + (size_t)s * (B * G3) + n;
#pragma unroll
  for (int r = 0; r < 8; ++r) {
    const int mm = r + (half << 3);
    orow[(size_t)mm * G3] = acc[r] + bias;
  }
}

// ---------------------------------------------------------------------------
// Kernel 2: persistent GRU scan (1365 sequential steps) + fused output
// projection. 512 threads = 16 waves, single workgroup. h lives in LDS
// (fp32 + bf16 shadow); x_proj is async-prefetched one step ahead into a
// double-buffered LDS staging area; W_hh/W_proj come from pre-packed bf16
// fragments (two b128 loads per fragment straight into WMMA).
// ---------------------------------------------------------------------------
__global__ void k2_gru_scan(const float* __restrict__ xproj,
                            const __bf16* __restrict__ whh_frag,
                            const float* __restrict__ b_hh,
                            const __bf16* __restrict__ wproj_frag,
                            const float* __restrict__ b_proj,
                            float* __restrict__ Ag) {
  constexpr int HSTR  = GH + 8;   // fp32 h stride (pad vs 64-bank LDS)
  constexpr int HBSTR = GH + 8;   // bf16 h stride (528 B rows, 16B aligned)
  constexpr int GSTR  = G3 + 8;
  __shared__ float  lds_xp[2][B * G3];     // 2 x 48 KB staging
  __shared__ float  lds_gh[B * GSTR];      // gate pre-activations
  __shared__ float  lds_h[B * HSTR];       // fp32 hidden state
  __shared__ __bf16 lds_hbf[B * HBSTR];    // bf16 shadow (WMMA A source)
  __shared__ float  lds_bhh[G3];
  __shared__ float  lds_bp[GH];

  const int tid  = threadIdx.x;
  const int wave = tid >> 5;
  const int lane = tid & 31;
  const int nl   = lane & 15;
  const int half = lane >> 4;

  for (int i = tid; i < B * HSTR; i += 512) lds_h[i] = 0.0f;
  for (int i = tid; i < B * HBSTR; i += 512) lds_hbf[i] = bfc(0.0f);
  for (int i = tid; i < G3; i += 512) lds_bhh[i] = b_hh[i];
  if (tid < GH) lds_bp[tid] = b_proj[tid];

  prefetch_xp(xproj, &lds_xp[0][0], tid);   // stage s = 0
  __syncthreads();

  for (int s = 0; s < NS; ++s) {
    // Stage s+1 while we compute (clamped re-read on the last step).
    const int snext = (s + 1 < NS) ? (s + 1) : (NS - 1);
    prefetch_xp(xproj + (size_t)snext * (B * G3), &lds_xp[(s + 1) & 1][0], tid);

    // -------- P1: gh = h @ W_hh^T (3 N-tiles per wave, K = 256) ------
    v8f acc0 = {}, acc1 = {}, acc2 = {};
    const __bf16* hrow = &lds_hbf[nl * HBSTR];
#pragma unroll
    for (int kc = 0; kc < 8; ++kc) {
      const int k0 = kc * 32;
      v16bf a  = load_a_row_bf(hrow, k0, half);
      v16bf b0 = load_b_packed(whh_frag, (wave * 3 + 0) * 8 + kc, lane);
      acc0 = WMMA_BF16(a, b0, acc0);
      v16bf b1 = load_b_packed(whh_frag, (wave * 3 + 1) * 8 + kc, lane);
      acc1 = WMMA_BF16(a, b1, acc1);
      v16bf b2 = load_b_packed(whh_frag, (wave * 3 + 2) * 8 + kc, lane);
      acc2 = WMMA_BF16(a, b2, acc2);
    }
#pragma unroll
    for (int r = 0; r < 8; ++r) {
      const int mm = r + (half << 3);
      lds_gh[mm * GSTR + (wave * 3 + 0) * 16 + nl] = acc0[r];
      lds_gh[mm * GSTR + (wave * 3 + 1) * 16 + nl] = acc1[r];
      lds_gh[mm * GSTR + (wave * 3 + 2) * 16 + nl] = acc2[r];
    }
    wait_async_keep6();          // buffer for step s is now resident
    __syncthreads();

    // -------- P2: gating -> h_new (fp32 + bf16 shadow) ---------------
    const float* xp = &lds_xp[s & 1][0];
#pragma unroll
    for (int i = 0; i < (B * GH) / 512; ++i) {
      const int e  = tid + 512 * i;
      const int mm = e >> 8;
      const int j  = e & 255;
      const float ghr = lds_gh[mm * GSTR + j]          + lds_bhh[j];
      const float ghz = lds_gh[mm * GSTR + GH + j]     + lds_bhh[GH + j];
      const float ghn = lds_gh[mm * GSTR + 2 * GH + j] + lds_bhh[2 * GH + j];
      const float xr = xp[mm * G3 + j];
      const float xz = xp[mm * G3 + GH + j];
      const float xn = xp[mm * G3 + 2 * GH + j];
      const float r  = sigf(xr + ghr);
      const float z  = sigf(xz + ghz);
      const float nn = tanhf(xn + r * ghn);
      const float hn = (1.0f - z) * nn + z * lds_h[mm * HSTR + j];
      lds_h[mm * HSTR + j]    = hn;
      lds_hbf[mm * HBSTR + j] = bfc(hn);
    }
    __syncthreads();

    // -------- P3: eps = h @ W_proj^T + b_proj; sigmoid on A half -----
    v8f ace = {};
#pragma unroll
    for (int kc = 0; kc < 8; ++kc) {
      v16bf a  = load_a_row_bf(hrow, kc * 32, half);
      v16bf bw = load_b_packed(wproj_frag, wave * 8 + kc, lane);
      ace = WMMA_BF16(a, bw, ace);
    }
    const int   n    = wave * 16 + nl;
    const float bias = lds_bp[n];
    float* arow = Ag + (size_t)s * (B * GH) + n;
#pragma unroll
    for (int r = 0; r < 8; ++r) {
      const int mm = r + (half << 3);
      float v = ace[r] + bias;
      if (n < H) v = sigf(v);
      arow[(size_t)mm * GH] = v;
    }
    // No trailing barrier needed: next P1 only reads lds_hbf / writes
    // lds_gh; the barrier after next P1 orders these reads vs P2 writes.
  }
}

// ---------------------------------------------------------------------------
// Kernel 3: fast diagonal SSM, one wave32 per fast frame (65552 frames).
// ---------------------------------------------------------------------------
__global__ void k3_fast_ssm(const float* __restrict__ x,
                            const float* __restrict__ Ag,
                            const float* __restrict__ W_in,
                            const float* __restrict__ W_out,
                            const float* __restrict__ Dp,
                            float* __restrict__ frames) {
  const int w = blockIdx.x * 8 + (threadIdx.x >> 5);  // exact grid
  const int lane = threadIdx.x & 31;
  const int b = w / NF;
  const int f = w - b * NF;
  int sidx = f / DELTA - 1;
  if (sidx < 0) sidx = 0;

  const float* ag = Ag + (size_t)sidx * (B * GH) + (size_t)b * GH;
  const float4 A4 = *(const float4*)(ag + lane * 4);
  const float4 g4 = *(const float4*)(ag + H + lane * 4);
  const float4 wi = *(const float4*)(W_in + lane * 4);
  const float4 wo = *(const float4*)(W_out + lane * 4);
  const float  D  = Dp[0];
  const float  xv = x[(size_t)b * T + (size_t)f * HOPF + lane];

  float4 h = {0.0f, 0.0f, 0.0f, 0.0f};
  float yout = 0.0f;
#pragma unroll
  for (int t = 0; t < LF; ++t) {
    const float xt = __shfl(xv, t, 32);
    h.x = A4.x * h.x + xt * wi.x;
    h.y = A4.y * h.y + xt * wi.y;
    h.z = A4.z * h.z + xt * wi.z;
    h.w = A4.w * h.w + xt * wi.w;
    float p = h.x * g4.x * wo.x + h.y * g4.y * wo.y +
              h.z * g4.z * wo.z + h.w * g4.w * wo.w;
#pragma unroll
    for (int off = 16; off > 0; off >>= 1) p += __shfl_xor(p, off, 32);
    if (lane == t) yout = p + D * xt;
  }
  frames[(size_t)w * LF + lane] = yout;
}

// ---------------------------------------------------------------------------
// Kernel 4: overlap-add gather (deterministic, no atomics).
// ---------------------------------------------------------------------------
__global__ void k4_overlap_add(const float* __restrict__ frames,
                               float* __restrict__ out) {
  const int gid = blockIdx.x * 256 + threadIdx.x;
  if (gid >= B * TOUT) return;
  const int b = gid / TOUT;
  const int t = gid - b * TOUT;
  const int f1 = t >> 4;
  const int o  = t & 15;
  float acc = 0.0f;
  if (f1 < NF) acc += frames[((size_t)b * NF + f1) * LF + o];
  if (f1 >= 1) acc += frames[((size_t)b * NF + f1 - 1) * LF + o + 16];
  out[gid] = acc;
}

// ---------------------------------------------------------------------------
extern "C" void kernel_launch(void* const* d_in, const int* in_sizes, int n_in,
                              void* d_out, int out_size, void* d_ws, size_t ws_size,
                              hipStream_t stream) {
  (void)in_sizes; (void)n_in; (void)out_size; (void)ws_size;

  const float* x      = (const float*)d_in[0];
  const float* W_ih   = (const float*)d_in[1];
  const float* W_hh   = (const float*)d_in[2];
  const float* b_ih   = (const float*)d_in[3];
  const float* b_hh   = (const float*)d_in[4];
  const float* W_proj = (const float*)d_in[5];
  const float* b_proj = (const float*)d_in[6];
  const float* W_in   = (const float*)d_in[7];
  const float* W_out  = (const float*)d_in[8];
  const float* Dp     = (const float*)d_in[9];
  float* out = (float*)d_out;

  // Workspace layout (all 32B aligned):
  //   xproj      : NS*16*768 fp32   (~67.1 MB)
  //   Ag         : NS*16*256 fp32   (~22.4 MB)
  //   frames     : 16*NF*32  fp32   (~8.4 MB)
  //   whh_frag   : 48*8*32*16 bf16  (384 KB)
  //   wproj_frag : 16*8*32*16 bf16  (128 KB)
  float*  xproj      = (float*)d_ws;
  float*  Ag         = xproj + (size_t)NS * B * G3;
  float*  frames     = Ag + (size_t)NS * B * GH;
  __bf16* whh_frag   = (__bf16*)(frames + (size_t)B * NF * LF);
  __bf16* wproj_frag = whh_frag + (size_t)48 * 8 * 32 * 16;

  constexpr int NFRAG_HH = 48 * 8;   // 384 fragments
  constexpr int NFRAG_PJ = 16 * 8;   // 128 fragments

  k0_pack_wfrag<<<(NFRAG_HH * 32 + 255) / 256, 256, 0, stream>>>(W_hh, GH, whh_frag, NFRAG_HH);
  k0_pack_wfrag<<<(NFRAG_PJ * 32 + 255) / 256, 256, 0, stream>>>(W_proj, GH, wproj_frag, NFRAG_PJ);
  k1_slow_proj<<<dim3(6, NS), 256, 0, stream>>>(x, W_ih, b_ih, xproj);
  k2_gru_scan<<<1, 512, 0, stream>>>(xproj, whh_frag, b_hh, wproj_frag, b_proj, Ag);
  k3_fast_ssm<<<(B * NF) / 8, 256, 0, stream>>>(x, Ag, W_in, W_out, Dp, frames);
  k4_overlap_add<<<(B * TOUT + 255) / 256, 256, 0, stream>>>(frames, out);
}
